// LengthRegulator_6124623364136
// MI455X (gfx1250) — compile-verified, matrix operand order checked by reading, not run on hardware
//
#include <hip/hip_runtime.h>

typedef __attribute__((ext_vector_type(16))) _Float16 v16h;
typedef __attribute__((ext_vector_type(8)))  float    v8f;
typedef __attribute__((ext_vector_type(4)))  float    v4f;
typedef __attribute__((ext_vector_type(4)))  int      v4i;

namespace {
constexpr int kB    = 16;
constexpr int kDim  = 384;
constexpr int kT    = 512;
constexpr int kDMax = 4096;            // T * 8 (static per reference)

constexpr int kTPB2 = 256;             // gather kernel block size (8 waves)
constexpr int kFVec = 4;               // frames per thread (float4 stores)
constexpr int kFramesPerBlock = kTPB2 * kFVec;   // 1024
constexpr int kDPB  = 16;              // dim-rows per block
}

// ---------------------------------------------------------------------------
// Phase 1: per-batch duration scan + scatter-style searchsorted inversion.
// One block (16 waves) per batch.
//   - wave32 shuffle inclusive scan of durations
//   - cross-wave 16-element prefix computed on the matrix unit:
//       p = L * s  done as  D = A x B,  A[m][k] = (k<=m) ? s[k] : 0,
//       B = all-ones (layout-invariant).  Exact: s[k] <= 224 (f16-exact),
//       f32 accumulate, totals <= 3584.
//   - thread t writes idx[f] = t for f in [cum_excl, cum_incl)  (disjoint)
// ---------------------------------------------------------------------------
__global__ __launch_bounds__(kT) void lr_scan_scatter(
    const int* __restrict__ duration,
    int*       __restrict__ idx,
    int*       __restrict__ mel_ws,
    int*       __restrict__ mel_out)
{
    __shared__ float s_wsum[16];
    __shared__ float s_pref[16];

    const int b    = blockIdx.x;
    const int t    = threadIdx.x;
    const int lane = t & 31;
    const int wave = t >> 5;

    int d = duration[b * kT + t];

    // wave32 inclusive scan
    int v = d;
    #pragma unroll
    for (int off = 1; off < 32; off <<= 1) {
        int n = __shfl_up(v, off, 32);
        if (lane >= off) v += n;
    }
    if (lane == 31) s_wsum[wave] = (float)v;
    __syncthreads();

    if (wave == 0) {
        // A-matrix (16x32 f16): lane m<16 holds row m, K = 0..7 in elems 0..7;
        // lane m+16 holds row m, K = 8..15 in elems 0..7; elems 8..15 are
        // K>=16 and stay zero.  A[m][k] = (k<=m) ? s[k] : 0.
        const int m  = lane & 15;
        const int kb = (lane < 16) ? 0 : 8;
        v16h a;
        #pragma unroll
        for (int e = 0; e < 8; ++e) {
            const int k = kb + e;
            a[e] = (k <= m) ? (_Float16)s_wsum[k] : (_Float16)0.0f;
        }
        #pragma unroll
        for (int e = 8; e < 16; ++e) a[e] = (_Float16)0.0f;

        v16h ones;
        #pragma unroll
        for (int e = 0; e < 16; ++e) ones[e] = (_Float16)1.0f;

        v8f c = {0.f, 0.f, 0.f, 0.f, 0.f, 0.f, 0.f, 0.f};
        c = __builtin_amdgcn_wmma_f32_16x16x32_f16(
                /*neg_a=*/false, a, /*neg_b=*/false, ones,
                /*c_mod=*/(short)0, c, /*reuse_a=*/false, /*reuse_b=*/false);

        // D layout: lanes 0-15 hold rows M=vgpr (p[0..7]); lanes 16-31 hold
        // rows M=vgpr+8 (p[8..15]).  Every column is identical, so lane 0 /
        // lane 16 suffice.
        if (lane == 0) {
            #pragma unroll
            for (int r = 0; r < 8; ++r) s_pref[r] = c[r];
        }
        if (lane == 16) {
            #pragma unroll
            for (int r = 0; r < 8; ++r) s_pref[8 + r] = c[r];
        }
    }
    __syncthreads();

    const int total = (int)s_pref[15];
    int cum = ((wave == 0) ? 0 : (int)s_pref[wave - 1]) + v;   // inclusive
    if (total == 0) { cum = t + 1; d = 1; }                    // row_sum==0 fallback
    const int cumx = cum - d;                                  // exclusive

    if (t == kT - 1) {
        mel_ws[b]  = cum;
        mel_out[b] = cum;      // mel_len written as raw int32 into d_out tail
    }

    int* row = idx + b * kDMax;
    for (int k = 0; k < d; ++k) row[cumx + k] = t;             // d in [0,7]
}

// ---------------------------------------------------------------------------
// Phase 2: expansion gather.  Bound by the 100.7 MB output stream.
// Threads map to contiguous frames -> coalesced 128-bit non-temporal stores.
// Each thread resolves its 4 indices once, then loops 16 dim-rows.
// ---------------------------------------------------------------------------
__global__ __launch_bounds__(kTPB2) void lr_gather(
    const float* __restrict__ x,
    const int*   __restrict__ idx,
    const int*   __restrict__ mel_ws,
    float*       __restrict__ out)
{
    const int b  = blockIdx.z;
    const int d0 = blockIdx.y * kDPB;
    const int f0 = (blockIdx.x * kTPB2 + (int)threadIdx.x) * kFVec;

    const int mel = mel_ws[b];

    v4i t4 = *(const v4i*)(idx + (size_t)b * kDMax + f0);

    int  tt[kFVec];
    bool va[kFVec];
    #pragma unroll
    for (int k = 0; k < kFVec; ++k) {
        int ti = t4[k];                       // may be poison when f>=mel
        ti = ti < 0 ? 0 : (ti > kT - 1 ? kT - 1 : ti);   // clamp => safe loads
        tt[k] = ti;
        va[k] = (f0 + k) < mel;
    }

    const float* xb = x   + ((size_t)b * kDim + d0) * (size_t)kT;
    float*       ob = out + ((size_t)b * kDim + d0) * (size_t)kDMax + f0;

    #pragma unroll
    for (int dd = 0; dd < kDPB; ++dd) {
        const float* xr = xb + (size_t)dd * kT;
        v4f r;
        #pragma unroll
        for (int k = 0; k < kFVec; ++k)
            r[k] = va[k] ? xr[tt[k]] : 0.0f;
        __builtin_nontemporal_store(r, (v4f*)(ob + (size_t)dd * kDMax));
    }
}

// ---------------------------------------------------------------------------
extern "C" void kernel_launch(void* const* d_in, const int* in_sizes, int n_in,
                              void* d_out, int out_size, void* d_ws, size_t ws_size,
                              hipStream_t stream) {
    (void)in_sizes; (void)n_in; (void)out_size; (void)ws_size;

    const float* x   = (const float*)d_in[0];
    const int*   dur = (const int*)d_in[1];
    // d_in[2] is d_max (device scalar); static value 4096 is baked in.

    float* out     = (float*)d_out;
    int*   idx     = (int*)d_ws;                         // B * d_max ints (256 KB)
    int*   mel_ws  = idx + (size_t)kB * kDMax;           // B ints
    int*   mel_out = (int*)d_out + (size_t)kB * kDim * kDMax;  // tuple tail

    lr_scan_scatter<<<kB, kT, 0, stream>>>(dur, idx, mel_ws, mel_out);

    dim3 g2(kDMax / kFramesPerBlock, kDim / kDPB, kB);   // 4 x 24 x 16 = 1536 blocks
    lr_gather<<<g2, kTPB2, 0, stream>>>(x, idx, mel_ws, out);
}
